// MultiHeadAttention_22720376995899
// MI455X (gfx1250) — matrix operand order repaired and checked by reading, not compiled, not run on hardware
//
#include <hip/hip_runtime.h>
#include <hip/hip_bf16.h>

typedef __bf16 bf16;
typedef __attribute__((ext_vector_type(16))) __bf16 v16bf;
typedef __attribute__((ext_vector_type(8)))  float  v8f;
typedef __attribute__((ext_vector_type(4)))  int    v4i;

#define SEQ   2048
#define BATCH 2
#define EMBED 2048
#define NHEAD 32
#define NKVH  8
#define HDIM  64
#define QKVD  3072   // EMBED + 2*512
#define MTOT  4096   // BATCH*SEQ

// ---------------------------------------------------------------------------
// CDNA5 async memory->LDS path (GLOBAL_LOAD_ASYNC_TO_LDS_B128, ASYNCcnt).
// Builtin signature (probe-verified): (as1 v4i*, as3 v4i*, imm, imm).
// ---------------------------------------------------------------------------
#if __has_builtin(__builtin_amdgcn_global_load_async_to_lds_b128) && \
    __has_builtin(__builtin_amdgcn_s_wait_asynccnt)
#define HAVE_ASYNC_LDS 1
typedef __attribute__((address_space(1))) v4i gv4i;   // int4 in global AS
typedef __attribute__((address_space(3))) v4i lv4i;   // int4 in LDS AS
#endif

__device__ __forceinline__ void async_copy16(const bf16* g, bf16* l) {
#ifdef HAVE_ASYNC_LDS
    // flat LDS pointer: low 32 bits == LDS byte offset (aperture truncation)
    __builtin_amdgcn_global_load_async_to_lds_b128(
        (gv4i*)(uintptr_t)g,
        (lv4i*)(uintptr_t)(unsigned)(uintptr_t)l, 0, 0);
#else
    *(uint4*)l = *(const uint4*)g;
#endif
}

template<int N> __device__ __forceinline__ void wait_async() {
#ifdef HAVE_ASYNC_LDS
    __builtin_amdgcn_s_wait_asynccnt(N);
#endif
}

// ---------------------------------------------------------------------------
// fp32 -> bf16 convert, 4 elements/thread (v_cvt_pk_bf16_f32 pairs)
// ---------------------------------------------------------------------------
__global__ void cvt_f32_bf16(const float* __restrict__ in, bf16* __restrict__ out, int n4) {
    int i = blockIdx.x * blockDim.x + threadIdx.x;
    if (i < n4) {
        float4 f = ((const float4*)in)[i];
        bf16 o4[4] = {(bf16)f.x, (bf16)f.y, (bf16)f.z, (bf16)f.w};
        *(uint2*)&out[(size_t)i * 4] = *(uint2*)o4;
    }
}

// ---------------------------------------------------------------------------
// C[M][N] (f32) = A[M][K] (bf16, row-major) * B[N][K]^T (bf16, row-major)
// 128x128 block tile, 4 waves, each wave 64(M) x 64(N): 16 WMMA per k-step
// against 16 ds_load_b128 (1:1). Double-buffered async LDS staging.
// ---------------------------------------------------------------------------
__global__ __launch_bounds__(128) void gemm_bf16_nt(
    const bf16* __restrict__ A, const bf16* __restrict__ B, float* __restrict__ C,
    int M, int N, int K)
{
    __shared__ bf16 As[2][128 * 40];   // padded stride 40 halves (80B)
    __shared__ bf16 Bs[2][128 * 40];

    const int tid  = threadIdx.x;
    const int lane = tid & 31;
    const int wave = tid >> 5;      // 0..3
    const int hl   = lane >> 4;
    const int r    = lane & 15;
    const int wm   = wave >> 1;     // 0..1 (M tile of 64)
    const int wn   = wave & 1;      // 0..1 (N tile of 64)
    const int m0   = blockIdx.y * 128;
    const int n0   = blockIdx.x * 128;

    v8f acc[4][4];
    #pragma unroll
    for (int mi = 0; mi < 4; ++mi)
        #pragma unroll
        for (int ni = 0; ni < 4; ++ni)
            acc[mi][ni] = (v8f){0.f,0.f,0.f,0.f,0.f,0.f,0.f,0.f};

    auto issue = [&](int buf, int kb) {
        #pragma unroll
        for (int t = 0; t < 4; ++t) {          // 8 async B128 per thread/stage
            int c   = tid + t * 128;
            int row = c >> 2;
            int seg = c & 3;
            async_copy16(&A[(size_t)(m0 + row) * K + kb + seg * 8],
                         &As[buf][row * 40 + seg * 8]);
            async_copy16(&B[(size_t)(n0 + row) * K + kb + seg * 8],
                         &Bs[buf][row * 40 + seg * 8]);
        }
    };

    const int nsteps = K >> 5;
    issue(0, 0);
    for (int it = 0; it < nsteps; ++it) {
        const int buf = it & 1;
        if (it + 1 < nsteps) { issue(buf ^ 1, (it + 1) * 32); wait_async<8>(); }
        else                 { wait_async<0>(); }
        __syncthreads();

        v16bf bfrag[4];
        #pragma unroll
        for (int ni = 0; ni < 4; ++ni) {
            int col = wn * 64 + ni * 16 + r;
            ((uint4*)&bfrag[ni])[0] = *(const uint4*)&Bs[buf][col * 40 + hl * 8];
            ((uint4*)&bfrag[ni])[1] = *(const uint4*)&Bs[buf][col * 40 + 16 + hl * 8];
        }
        #pragma unroll
        for (int mi = 0; mi < 4; ++mi) {
            int row = wm * 64 + mi * 16 + r;
            v16bf afrag;
            ((uint4*)&afrag)[0] = *(const uint4*)&As[buf][row * 40 + hl * 8];
            ((uint4*)&afrag)[1] = *(const uint4*)&As[buf][row * 40 + 16 + hl * 8];
            #pragma unroll
            for (int ni = 0; ni < 4; ++ni)
                acc[mi][ni] = __builtin_amdgcn_wmma_f32_16x16x32_bf16(
                    false, afrag, false, bfrag[ni], (short)0, acc[mi][ni], false, false);
        }
        __syncthreads();
    }

    #pragma unroll
    for (int mi = 0; mi < 4; ++mi)
        #pragma unroll
        for (int ni = 0; ni < 4; ++ni)
            #pragma unroll
            for (int i = 0; i < 8; ++i) {
                int row = m0 + wm * 64 + mi * 16 + i + hl * 8;
                int col = n0 + wn * 64 + ni * 16 + r;
                C[(size_t)row * N + col] = acc[mi][ni][i];
            }
}

// ---------------------------------------------------------------------------
// Per-head LayerNorm of Q,K (Dh=64) + bf16 split of qkv, V transposed.
// roles: 0..31 q, 32..39 k, 40..47 v.
// ---------------------------------------------------------------------------
__global__ __launch_bounds__(256) void qkv_norm_split(
    const float* __restrict__ qkv, const float* __restrict__ nw, const float* __restrict__ nb,
    bf16* __restrict__ qn, bf16* __restrict__ kn, bf16* __restrict__ vt)
{
    int task = blockIdx.x * 8 + (threadIdx.x >> 5);
    int lane = threadIdx.x & 31;
    int t    = task / 48;
    int role = task % 48;
    int b    = t >> 11;
    int l    = t & 2047;
    int d0 = lane, d1 = lane + 32;

    const float* src;
    if (role < 32)       src = qkv + (size_t)t * QKVD + role * 64;
    else if (role < 40)  src = qkv + (size_t)t * QKVD + 2048 + (role - 32) * 64;
    else                 src = qkv + (size_t)t * QKVD + 2560 + (role - 40) * 64;

    float x0 = src[d0], x1 = src[d1];

    if (role < 40) {
        float s = x0 + x1;
        #pragma unroll
        for (int m = 16; m >= 1; m >>= 1) s += __shfl_xor(s, m, 32);
        float mu = s * (1.0f / 64.0f);
        float vv = (x0 - mu) * (x0 - mu) + (x1 - mu) * (x1 - mu);
        #pragma unroll
        for (int m = 16; m >= 1; m >>= 1) vv += __shfl_xor(vv, m, 32);
        float rstd = rsqrtf(vv * (1.0f / 64.0f) + 1e-8f);
        float y0 = (x0 - mu) * rstd * nw[d0] + nb[d0];
        float y1 = (x1 - mu) * rstd * nw[d1] + nb[d1];
        bf16* dst = (role < 32)
            ? qn + ((size_t)(b * NHEAD + role)        * SEQ + l) * HDIM
            : kn + ((size_t)(b * NKVH  + (role - 32)) * SEQ + l) * HDIM;
        dst[d0] = (bf16)y0;
        dst[d1] = (bf16)y1;
    } else {
        int kh = role - 40;
        bf16* dst = vt + (size_t)(b * NKVH + kh) * HDIM * SEQ;
        dst[(size_t)d0 * SEQ + l] = (bf16)x0;
        dst[(size_t)d1 * SEQ + l] = (bf16)x1;
    }
}

// ---------------------------------------------------------------------------
// Flash attention: block per (b, h, 64 q-rows); 4 waves x 16 q-rows.
// Key loop 32 keys/step, double-buffered async K/V staging.
// ---------------------------------------------------------------------------
__global__ __launch_bounds__(128) void flash_attn(
    const bf16* __restrict__ qn, const bf16* __restrict__ kn,
    const bf16* __restrict__ vt, bf16* __restrict__ o)
{
    __shared__ bf16 Ks[2][32 * 72];    // [key][d]
    __shared__ bf16 Vs[2][64 * 40];    // [d][key]
    __shared__ bf16 Ps[4][16 * 40];    // per-wave P tile

    const int tid  = threadIdx.x;
    const int lane = tid & 31;
    const int wave = tid >> 5;
    const int hl   = lane >> 4;
    const int r    = lane & 15;
    const int qt   = blockIdx.x;
    const int h    = blockIdx.y;
    const int b    = blockIdx.z;
    const int kh   = h >> 2;

    const bf16* qbase = qn + (size_t)(b * NHEAD + h)  * SEQ * HDIM;
    const bf16* kbase = kn + (size_t)(b * NKVH + kh)  * SEQ * HDIM;
    const bf16* vbase = vt + (size_t)(b * NKVH + kh)  * HDIM * SEQ;

    const int qrow = qt * 64 + wave * 16 + r;
    v16bf qf[2];
    #pragma unroll
    for (int ks = 0; ks < 2; ++ks) {
        ((uint4*)&qf[ks])[0] = *(const uint4*)&qbase[(size_t)qrow * HDIM + ks * 32 + hl * 8];
        ((uint4*)&qf[ks])[1] = *(const uint4*)&qbase[(size_t)qrow * HDIM + ks * 32 + 16 + hl * 8];
    }

    float mrow[8], lrow[8];
    v8f accO[4];
    #pragma unroll
    for (int i = 0; i < 8; ++i) { mrow[i] = -3.0e38f; lrow[i] = 0.f; }
    #pragma unroll
    for (int di = 0; di < 4; ++di) accO[di] = (v8f){0.f,0.f,0.f,0.f,0.f,0.f,0.f,0.f};

    auto issue = [&](int buf, int kb) {
        #pragma unroll
        for (int t = 0; t < 2; ++t) {           // K: 32 x 64 halves
            int c = tid + t * 128;
            int row = c >> 3, seg = c & 7;
            async_copy16(&kbase[(size_t)(kb + row) * HDIM + seg * 8],
                         &Ks[buf][row * 72 + seg * 8]);
        }
        #pragma unroll
        for (int t = 0; t < 2; ++t) {           // V^T: 64 x 32 halves
            int c = tid + t * 128;
            int row = c >> 2, seg = c & 3;
            async_copy16(&vbase[(size_t)row * SEQ + kb + seg * 8],
                         &Vs[buf][row * 40 + seg * 8]);
        }
    };

    const int nsteps = SEQ >> 5;   // 64
    issue(0, 0);
    for (int it = 0; it < nsteps; ++it) {
        const int buf = it & 1;
        if (it + 1 < nsteps) { issue(buf ^ 1, (it + 1) * 32); wait_async<4>(); }
        else                 { wait_async<0>(); }
        __syncthreads();

        // S(16x32) = Q(16x64) * K^T(64x32)
        v8f s[2];
        s[0] = (v8f){0.f,0.f,0.f,0.f,0.f,0.f,0.f,0.f};
        s[1] = (v8f){0.f,0.f,0.f,0.f,0.f,0.f,0.f,0.f};
        #pragma unroll
        for (int ni = 0; ni < 2; ++ni) {
            int key = ni * 16 + r;
            #pragma unroll
            for (int ks = 0; ks < 2; ++ks) {
                v16bf kf;
                ((uint4*)&kf)[0] = *(const uint4*)&Ks[buf][key * 72 + ks * 32 + hl * 8];
                ((uint4*)&kf)[1] = *(const uint4*)&Ks[buf][key * 72 + ks * 32 + 16 + hl * 8];
                s[ni] = __builtin_amdgcn_wmma_f32_16x16x32_bf16(
                    false, qf[ks], false, kf, (short)0, s[ni], false, false);
            }
        }

        // online softmax per C slot (row = slot + 8*half)
        #pragma unroll
        for (int i = 0; i < 8; ++i) {
            float mx = fmaxf(s[0][i], s[1][i]);
            #pragma unroll
            for (int m = 8; m >= 1; m >>= 1) mx = fmaxf(mx, __shfl_xor(mx, m, 32));
            float mnew  = fmaxf(mrow[i], mx);
            float scale = __expf(mrow[i] - mnew);
            float p0 = __expf(s[0][i] - mnew);
            float p1 = __expf(s[1][i] - mnew);
            float ps = p0 + p1;
            #pragma unroll
            for (int m = 8; m >= 1; m >>= 1) ps += __shfl_xor(ps, m, 32);
            lrow[i] = lrow[i] * scale + ps;
            mrow[i] = mnew;
            #pragma unroll
            for (int di = 0; di < 4; ++di) accO[di][i] *= scale;
            Ps[wave][(i + hl * 8) * 40 + r]      = (bf16)p0;
            Ps[wave][(i + hl * 8) * 40 + 16 + r] = (bf16)p1;
        }

        v16bf pf;
        ((uint4*)&pf)[0] = *(const uint4*)&Ps[wave][r * 40 + hl * 8];
        ((uint4*)&pf)[1] = *(const uint4*)&Ps[wave][r * 40 + 16 + hl * 8];

        // O(16x64) += P(16x32) * V(32x64)
        #pragma unroll
        for (int di = 0; di < 4; ++di) {
            v16bf vf;
            ((uint4*)&vf)[0] = *(const uint4*)&Vs[buf][(di * 16 + r) * 40 + hl * 8];
            ((uint4*)&vf)[1] = *(const uint4*)&Vs[buf][(di * 16 + r) * 40 + 16 + hl * 8];
            accO[di] = __builtin_amdgcn_wmma_f32_16x16x32_bf16(
                false, pf, false, vf, (short)0, accO[di], false, false);
        }
        __syncthreads();
    }

    #pragma unroll
    for (int di = 0; di < 4; ++di)
        #pragma unroll
        for (int i = 0; i < 8; ++i) {
            float ov = accO[di][i] / lrow[i];
            int row  = qt * 64 + wave * 16 + i + hl * 8;
            o[((size_t)(b * SEQ + row)) * EMBED + h * HDIM + di * 16 + r] = (bf16)ov;
        }
}

// ---------------------------------------------------------------------------
// launch
// ---------------------------------------------------------------------------
extern "C" void kernel_launch(void* const* d_in, const int* in_sizes, int n_in,
                              void* d_out, int out_size, void* d_ws, size_t ws_size,
                              hipStream_t stream)
{
    const float* x     = (const float*)d_in[0];
    const float* w_in  = (const float*)d_in[1];
    const float* w_out = (const float*)d_in[2];
    const float* nw    = (const float*)d_in[3];
    const float* nb    = (const float*)d_in[4];
    float* out = (float*)d_out;

    char* ws = (char*)d_ws;
    bf16*  x_b    = (bf16*)(ws + 0);            //  16.78 MB
    bf16*  win_b  = (bf16*)(ws + 16777216);     //  12.58 MB
    bf16*  wout_b = (bf16*)(ws + 29360128);     //   8.39 MB
    float* qkv    = (float*)(ws + 37748736);    //  50.33 MB
    bf16*  qn     = (bf16*)(ws + 88080384);     //  16.78 MB
    bf16*  kn     = (bf16*)(ws + 104857600);    //   4.19 MB
    bf16*  vt     = (bf16*)(ws + 109051904);    //   4.19 MB
    bf16*  ob     = (bf16*)(ws + 113246208);    //  16.78 MB  (total ~130 MB)

    cvt_f32_bf16<<<(MTOT * EMBED / 4 + 255) / 256, 256, 0, stream>>>(x, x_b, MTOT * EMBED / 4);
    cvt_f32_bf16<<<(QKVD * EMBED / 4 + 255) / 256, 256, 0, stream>>>(w_in, win_b, QKVD * EMBED / 4);
    cvt_f32_bf16<<<(EMBED * EMBED / 4 + 255) / 256, 256, 0, stream>>>(w_out, wout_b, EMBED * EMBED / 4);

    gemm_bf16_nt<<<dim3(QKVD / 128, MTOT / 128), 128, 0, stream>>>(
        x_b, win_b, qkv, MTOT, QKVD, EMBED);

    qkv_norm_split<<<(MTOT * 48) / 8, 256, 0, stream>>>(qkv, nw, nb, qn, kn, vt);

    flash_attn<<<dim3(SEQ / 64, NHEAD, BATCH), 128, 0, stream>>>(qn, kn, vt, ob);

    gemm_bf16_nt<<<dim3(EMBED / 128, MTOT / 128), 128, 0, stream>>>(
        ob, wout_b, out, MTOT, EMBED, EMBED);
}
